// DynGraphLayer_66640712564950
// MI455X (gfx1250) — matrix operand order repaired and checked by reading, not compiled, not run on hardware
//
#include <hip/hip_runtime.h>

// ---------------------------------------------------------------------------
// Types / helpers
// ---------------------------------------------------------------------------
typedef __attribute__((ext_vector_type(16))) __bf16 bf16x16;
typedef __attribute__((ext_vector_type(2)))  __bf16 bf16x2;
typedef __attribute__((ext_vector_type(8)))  float  f32x8;

union Frag {
    unsigned u[8];
    bf16x16  v;
};

__device__ __forceinline__ unsigned pack_bf16(float a, float b) {
#if __has_builtin(__builtin_amdgcn_cvt_pk_bf16_f32)
    bf16x2 p = __builtin_amdgcn_cvt_pk_bf16_f32(a, b);   // v_cvt_pk_bf16_f32
    return __builtin_bit_cast(unsigned, p);
#else
    unsigned ua = __builtin_bit_cast(unsigned, a);
    unsigned ub = __builtin_bit_cast(unsigned, b);
    ua = (ua + 0x7FFFu + ((ua >> 16) & 1u)) >> 16;       // RNE fp32 -> bf16
    ub = (ub + 0x7FFFu + ((ub >> 16) & 1u)) >> 16;
    return (ub << 16) | (ua & 0xFFFFu);
#endif
}

// ---------------------------------------------------------------------------
// Generic GEMM:  out[m, ocol0+n] = act( A[m,:K] @ W[:K, n0..] + bias[n] )
// A row-major, lda == K.  W row-major [K, 256].  Nc = 256 fixed.
// Block tile: 128 rows (8 waves x 16) x 64 cols (4 wmma tiles / wave),
// LDS K-tile = 64 (two wmma k-iterations per barrier pair).
// ---------------------------------------------------------------------------
__global__ __launch_bounds__(256) void gemm_bias_kernel(
    const float* __restrict__ A, const float* __restrict__ W,
    const float* __restrict__ bias, float* __restrict__ out,
    int K, int ldo, int ocol0, int relu)
{
    const int Nc = 256;
    __shared__ unsigned Alds[128][33];   // [row][kpair 0..31] bf16x2, +pad
    __shared__ unsigned Blds[64][33];    // [col][kpair 0..31] bf16x2, +pad

    const int tid  = threadIdx.x;
    const int lane = tid & 31;
    const int wave = tid >> 5;
    const int half = lane >> 4;
    const int lm   = lane & 15;

    const int m0 = blockIdx.x * 128;
    const int n0 = blockIdx.y * 64;

    f32x8 acc[4] = {};

    for (int k0 = 0; k0 < K; k0 += 64) {
        // ---- A tile 128x64 f32 -> bf16 pairs (2 threads per row) ----
        {
            int row = tid >> 1;
            int seg = (tid & 1) * 16;              // kpair base 0 or 16
            const float* src = A + (size_t)(m0 + row) * K + k0 + seg * 2;
            #pragma unroll
            for (int i = 0; i < 16; ++i)
                Alds[row][seg + i] = pack_bf16(src[2 * i], src[2 * i + 1]);
        }
        // ---- W tile 64x64 -> Blds[n][kpair] (pack along K) ----
        {
            #pragma unroll
            for (int it = 0; it < 8; ++it) {
                int f  = tid + it * 256;           // 0..2047
                int n  = f & 63;
                int kp = f >> 6;                   // 0..31
                const float* src = W + (size_t)(k0 + 2 * kp) * Nc + n0 + n;
                Blds[n][kp] = pack_bf16(src[0], src[Nc]);
            }
        }
        __syncthreads();

        // ---- two WMMA k-iterations out of the LDS tile ----
        #pragma unroll
        for (int ks = 0; ks < 2; ++ks) {
            Frag a;
            int arow = wave * 16 + lm;
            #pragma unroll
            for (int i = 0; i < 4; ++i) {
                a.u[i]     = Alds[arow][ks * 16 + half * 4 + i];
                a.u[4 + i] = Alds[arow][ks * 16 + 8 + half * 4 + i];
            }
            #pragma unroll
            for (int ct = 0; ct < 4; ++ct) {
                Frag bfr;
                int n = ct * 16 + lm;
                #pragma unroll
                for (int i = 0; i < 8; ++i) bfr.u[i] = Blds[n][ks * 16 + (i >> 2) * 8 + half * 4 + (i & 3)];
                acc[ct] = __builtin_amdgcn_wmma_f32_16x16x32_bf16(
                    false, a.v, false, bfr.v, (short)0, acc[ct], false, false);
            }
        }
        __syncthreads();
    }

    // ---- epilogue: bias (+relu), C layout: VGPR r -> M = r + 8*half ----
    #pragma unroll
    for (int ct = 0; ct < 4; ++ct) {
        int col = n0 + ct * 16 + lm;
        float bv = bias[col];
        #pragma unroll
        for (int r = 0; r < 8; ++r) {
            int row = m0 + wave * 16 + r + half * 8;
            float vv = acc[ct][r] + bv;
            if (relu) vv = fmaxf(vv, 0.0f);
            out[(size_t)row * ldo + ocol0 + col] = vv;
        }
    }
}

// ---------------------------------------------------------------------------
// Masked flash attention.  N=1024, H*d=256, d=64, scale=1/8.
// Block: (128 q-rows, head, batch), 8 waves, each wave owns 16 q-rows.
// ---------------------------------------------------------------------------
__global__ __launch_bounds__(256) void graph_attn_kernel(
    const float* __restrict__ Q, const float* __restrict__ Km,
    const float* __restrict__ Vm, const int* __restrict__ adj,
    float* __restrict__ out, int ldo, int ocol_base)
{
    const int N = 1024, HD = 256, D = 64;
    const int b  = blockIdx.z;
    const int h  = blockIdx.y;
    const int q0 = blockIdx.x * 128;

    const int tid  = threadIdx.x;
    const int lane = tid & 31;
    const int wave = tid >> 5;
    const int half = lane >> 4;
    const int lm   = lane & 15;

    __shared__ unsigned Kp[64][33];              // [key][d-pair 0..31]
    __shared__ unsigned Vp[64][33];              // [dd][key-pair 0..31]
    __shared__ unsigned short Plds[8][16][68];   // per-wave P, bf16 [row][key]

    // ---- Q fragments, resident in registers for the whole block ----
    Frag aq[2];
    {
        int row = q0 + wave * 16 + lm;
        const float* qp = Q + ((size_t)(b * N + row)) * HD + h * D;
        #pragma unroll
        for (int s = 0; s < 2; ++s) {
            int base = s * 32 + half * 8;
            #pragma unroll
            for (int i = 0; i < 4; ++i) {
                aq[s].u[i]     = pack_bf16(qp[base + 2 * i],      qp[base + 2 * i + 1]);
                aq[s].u[4 + i] = pack_bf16(qp[base + 16 + 2 * i], qp[base + 17 + 2 * i]);
            }
        }
    }

    f32x8 o[4] = {};
    float mrow[8], lrow[8];
    #pragma unroll
    for (int r = 0; r < 8; ++r) { mrow[r] = -INFINITY; lrow[r] = 0.0f; }

    for (int k0 = 0; k0 < N; k0 += 64) {
        // ---- K tile: pack along d -> Kp[key][dpair] ----
        {
            int key = tid >> 2;                    // 0..63
            int kpb = (tid & 3) * 8;               // dpair base
            const float* src = Km + ((size_t)(b * N + k0 + key)) * HD + h * D + kpb * 2;
            #pragma unroll
            for (int i = 0; i < 8; ++i)
                Kp[key][kpb + i] = pack_bf16(src[2 * i], src[2 * i + 1]);
        }
        // ---- V tile: transpose-pack along key -> Vp[dd][keypair] ----
        {
            #pragma unroll
            for (int it = 0; it < 8; ++it) {
                int f  = tid + it * 256;           // 0..2047
                int dd = f & 63;
                int kk = f >> 6;                   // 0..31
                const float* src = Vm + ((size_t)(b * N + k0 + 2 * kk)) * HD + h * D + dd;
                Vp[dd][kk] = pack_bf16(src[0], src[HD]);
            }
        }
        __syncthreads();

        // ---- S = Q @ K^T  (d=64 -> 2 wmma k-steps) ----
        f32x8 s[4];
        #pragma unroll
        for (int ct = 0; ct < 4; ++ct) {
            Frag b0, b1;
            int n = ct * 16 + lm;
            #pragma unroll
            for (int i = 0; i < 8; ++i) {
                b0.u[i] = Kp[n][(i >> 2) * 8 + half * 4 + (i & 3)];
                b1.u[i] = Kp[n][16 + (i >> 2) * 8 + half * 4 + (i & 3)];
            }
            f32x8 z = {};
            z = __builtin_amdgcn_wmma_f32_16x16x32_bf16(false, aq[0].v, false, b0.v, (short)0, z, false, false);
            z = __builtin_amdgcn_wmma_f32_16x16x32_bf16(false, aq[1].v, false, b1.v, (short)0, z, false, false);
            s[ct] = z;
        }

        // ---- scale + adjacency mask (-1e9 like the reference) ----
        #pragma unroll
        for (int ct = 0; ct < 4; ++ct) {
            int key = k0 + ct * 16 + lm;
            const int* ap = adj + ((size_t)(b * N + q0 + wave * 16)) * N + key;
            #pragma unroll
            for (int r = 0; r < 8; ++r) {
                int m = ap[(size_t)(r + half * 8) * N];
                s[ct][r] = m ? s[ct][r] * 0.125f : -1e9f;
            }
        }

        // ---- online softmax (row stats per 16-lane half via shfl_xor) ----
        float alpha[8];
        #pragma unroll
        for (int r = 0; r < 8; ++r) {
            float mx = fmaxf(fmaxf(s[0][r], s[1][r]), fmaxf(s[2][r], s[3][r]));
            #pragma unroll
            for (int off = 1; off < 16; off <<= 1) mx = fmaxf(mx, __shfl_xor(mx, off, 32));
            float mnew = fmaxf(mrow[r], mx);
            alpha[r] = __expf(mrow[r] - mnew);
            mrow[r]  = mnew;
            float sum = 0.0f;
            #pragma unroll
            for (int ct = 0; ct < 4; ++ct) {
                float p = __expf(s[ct][r] - mnew);
                s[ct][r] = p;
                sum += p;
            }
            #pragma unroll
            for (int off = 1; off < 16; off <<= 1) sum += __shfl_xor(sum, off, 32);
            lrow[r] = lrow[r] * alpha[r] + sum;
            #pragma unroll
            for (int ct = 0; ct < 4; ++ct) o[ct][r] *= alpha[r];
        }

        // ---- P -> per-wave LDS strip as bf16 (A-fragment re-read) ----
        #pragma unroll
        for (int ct = 0; ct < 4; ++ct) {
            #pragma unroll
            for (int r = 0; r < 8; ++r) {
                Plds[wave][r + half * 8][ct * 16 + lm] =
                    (unsigned short)(pack_bf16(s[ct][r], 0.0f) & 0xFFFFu);
            }
        }
        __syncthreads();

        // ---- O += P @ V  (key=64 -> 2 wmma k-steps) ----
        #pragma unroll
        for (int st = 0; st < 2; ++st) {
            Frag ap2;
            const unsigned short* prow = &Plds[wave][lm][0];
            #pragma unroll
            for (int i = 0; i < 4; ++i) {
                int kidx = st * 32 + half * 8 + 2 * i;        // even -> dword aligned
                ap2.u[i]     = *(const unsigned*)(prow + kidx);
                ap2.u[4 + i] = *(const unsigned*)(prow + kidx + 16);
            }
            #pragma unroll
            for (int ct = 0; ct < 4; ++ct) {
                Frag bv;
                int dd = ct * 16 + lm;
                #pragma unroll
                for (int i = 0; i < 8; ++i) bv.u[i] = Vp[dd][st * 16 + (i >> 2) * 8 + half * 4 + (i & 3)];
                o[ct] = __builtin_amdgcn_wmma_f32_16x16x32_bf16(
                    false, ap2.v, false, bv.v, (short)0, o[ct], false, false);
            }
        }
        __syncthreads();
    }

    // ---- normalize + write into concat destination ----
    #pragma unroll
    for (int ct = 0; ct < 4; ++ct) {
        int col = ocol_base + h * D + ct * 16 + lm;
        #pragma unroll
        for (int r = 0; r < 8; ++r) {
            int row = q0 + wave * 16 + r + half * 8;
            out[(size_t)(b * N + row) * ldo + col] = o[ct][r] / lrow[r];
        }
    }
}

// ---------------------------------------------------------------------------
// Strided column copy for the concatenations.
// ---------------------------------------------------------------------------
__global__ void copy_cols_kernel(const float* __restrict__ src, float* __restrict__ dst,
                                 int src_w, int dst_w, size_t total)
{
    size_t i = blockIdx.x * (size_t)blockDim.x + threadIdx.x;
    if (i < total) {
        size_t row = i / src_w, c = i % src_w;
        dst[row * dst_w + c] = src[i];
    }
}

// ---------------------------------------------------------------------------
// Orchestration
// ---------------------------------------------------------------------------
extern "C" void kernel_launch(void* const* d_in, const int* in_sizes, int n_in,
                              void* d_out, int out_size, void* d_ws, size_t ws_size,
                              hipStream_t stream)
{
    (void)in_sizes; (void)n_in; (void)out_size; (void)ws_size;

    const float* x    = (const float*)d_in[0];
    const int*   adj  = (const int*)d_in[1];
    const float* Wq0  = (const float*)d_in[2];
    const float* bq0  = (const float*)d_in[3];
    const float* Wk0  = (const float*)d_in[4];
    const float* bk0  = (const float*)d_in[5];
    const float* Wv0  = (const float*)d_in[6];
    const float* bv0  = (const float*)d_in[7];
    const float* Wq1  = (const float*)d_in[8];
    const float* bq1  = (const float*)d_in[9];
    const float* Wk1  = (const float*)d_in[10];
    const float* bk1  = (const float*)d_in[11];
    const float* Wv1  = (const float*)d_in[12];
    const float* bv1  = (const float*)d_in[13];
    const float* Wout = (const float*)d_in[14];
    const float* bout = (const float*)d_in[15];
    float* outp = (float*)d_out;

    // workspace layout (floats): q,k,v [16384,256]; x1 [16384,512]; x2 [16384,768]
    float* ws = (float*)d_ws;
    float* q  = ws;
    float* k  = ws + 4194304;
    float* v  = ws + 8388608;
    float* x1 = ws + 12582912;
    float* x2 = ws + 20971520;

    const dim3 gGemm(128, 4, 1);     // M=16384/128, Nc=256/64
    const dim3 gAttn(8, 4, 16);      // N/128, H, B
    const int  blk = 256;

    // --- iteration 0: projections (K = 256) ---
    gemm_bias_kernel<<<gGemm, blk, 0, stream>>>(x, Wq0, bq0, q, 256, 256, 0, 0);
    gemm_bias_kernel<<<gGemm, blk, 0, stream>>>(x, Wk0, bk0, k, 256, 256, 0, 0);
    gemm_bias_kernel<<<gGemm, blk, 0, stream>>>(x, Wv0, bv0, v, 256, 256, 0, 0);

    // x1[:, 0:256] = x ;  x1[:, 256:512] = attn0
    {
        size_t total = (size_t)16384 * 256;
        copy_cols_kernel<<<(unsigned)((total + blk - 1) / blk), blk, 0, stream>>>(x, x1, 256, 512, total);
    }
    graph_attn_kernel<<<gAttn, blk, 0, stream>>>(q, k, v, adj, x1, 512, 256);

    // --- iteration 1: projections (K = 512), reuse q/k/v workspace ---
    gemm_bias_kernel<<<gGemm, blk, 0, stream>>>(x1, Wq1, bq1, q, 512, 256, 0, 0);
    gemm_bias_kernel<<<gGemm, blk, 0, stream>>>(x1, Wk1, bk1, k, 512, 256, 0, 0);
    gemm_bias_kernel<<<gGemm, blk, 0, stream>>>(x1, Wv1, bv1, v, 512, 256, 0, 0);

    // x2[:, 0:512] = x1 ; x2[:, 512:768] = attn1
    {
        size_t total = (size_t)16384 * 512;
        copy_cols_kernel<<<(unsigned)((total + blk - 1) / blk), blk, 0, stream>>>(x1, x2, 512, 768, total);
    }
    graph_attn_kernel<<<gAttn, blk, 0, stream>>>(q, k, v, adj, x2, 768, 512);

    // --- output: relu(x2 @ Wout + bout) -> d_out (K = 768) ---
    gemm_bias_kernel<<<gGemm, blk, 0, stream>>>(x2, Wout, bout, outp, 768, 256, 0, 1);
}
